// TulparMambaBlock_36344013259239
// MI455X (gfx1250) — matrix operand order repaired and checked by reading, not compiled, not run on hardware
//
#include <hip/hip_runtime.h>
#include <hip/hip_bf16.h>

#define B_SZ 4
#define LSEQ 2048
#define DMODEL 1024
#define DSTATE 16
#define DOUT 1024

typedef __attribute__((ext_vector_type(16))) __bf16 v16bf;
typedef __attribute__((ext_vector_type(8)))  float  v8f;
typedef __attribute__((ext_vector_type(16))) unsigned short v16u;
typedef __attribute__((ext_vector_type(8)))  unsigned short v8u;

__device__ __forceinline__ unsigned short f2bf(float f) {
    unsigned int u = __float_as_uint(f);
    u += 0x7FFFu + ((u >> 16) & 1u);           // round-to-nearest-even
    return (unsigned short)(u >> 16);
}

// ---------------------------------------------------------------------------
// Kernel 1: transpose x (B,L,D) -> xT (B,D,L) so the scan streams contiguous l.
// ---------------------------------------------------------------------------
__global__ __launch_bounds__(256)
void k_transpose(const float* __restrict__ x, float* __restrict__ xT) {
    __shared__ float tile[32][33];
    const int d0 = blockIdx.x * 32;
    const int l0 = blockIdx.y * 32;
    const int b  = blockIdx.z;
    const int tx = threadIdx.x, ty = threadIdx.y;
#pragma unroll
    for (int i = 0; i < 32; i += 8)
        tile[ty + i][tx] = x[((size_t)b * LSEQ + (l0 + ty + i)) * DMODEL + d0 + tx];
    __syncthreads();
#pragma unroll
    for (int i = 0; i < 32; i += 8)
        xT[((size_t)b * DMODEL + (d0 + ty + i)) * LSEQ + l0 + tx] = tile[tx][ty + i];
}

// ---------------------------------------------------------------------------
// Kernel 2: BC[row,e] = sum_k x[row,k]*W_xp[e,k] + b_xp[e]   (row = b*L+l, e<32)
// ---------------------------------------------------------------------------
__global__ __launch_bounds__(256)
void k_bcproj(const float* __restrict__ x, const float* __restrict__ W_xp,
              const float* __restrict__ b_xp, float* __restrict__ BC) {
    __shared__ float red[8][32];
    const int row = blockIdx.x;
    const int e = threadIdx.x & 31;
    const int s = threadIdx.x >> 5;          // K-slice 0..7, 128 k each
    const float* xr = x + (size_t)row * DMODEL + s * 128;
    const float* wr = W_xp + (size_t)e * DMODEL + s * 128;
    float acc = 0.f;
#pragma unroll 4
    for (int i = 0; i < 128; i += 4) {
        float4 xv = *(const float4*)(xr + i);
        float4 wv = *(const float4*)(wr + i);
        acc = fmaf(xv.x, wv.x, acc); acc = fmaf(xv.y, wv.y, acc);
        acc = fmaf(xv.z, wv.z, acc); acc = fmaf(xv.w, wv.w, acc);
    }
    red[s][e] = acc;
    __syncthreads();
    if (threadIdx.x < 32) {
        float t = 0.f;
#pragma unroll
        for (int s2 = 0; s2 < 8; ++s2) t += red[s2][threadIdx.x];
        BC[(size_t)row * 32 + threadIdx.x] = t + b_xp[threadIdx.x];
    }
}

// ---------------------------------------------------------------------------
// Kernel 3: W_out f32 -> bf16
// ---------------------------------------------------------------------------
__global__ __launch_bounds__(256)
void k_wconv(const float* __restrict__ W, unsigned short* __restrict__ Wb, int n) {
    int i = blockIdx.x * blockDim.x + threadIdx.x;
    if (i < n) Wb[i] = f2bf(W[i]);
}

// ---------------------------------------------------------------------------
// Kernel 4: fused dt/softplus + selective scan.
// 16 lanes per d (one per state n); wave32 = 2 d's; block=512 covers 32 d's of
// one batch. Per 16-step chunk, lane j precomputes softplus-dt and u=dt*x for
// step j ONCE (instead of 16x redundantly), then broadcasts via shuffles. The
// serial inner step is: 2 bcast + BC load + 2 shfl + 1 exp + 2 fma + reduce.
// ---------------------------------------------------------------------------
__global__ __launch_bounds__(512)
void k_scan(const float* __restrict__ xT, const float* __restrict__ BC,
            const float* __restrict__ A_log, const float* __restrict__ W_dt,
            const float* __restrict__ b_dt, unsigned short* __restrict__ y_bf) {
    __shared__ float ytile[16][32];
    const int b  = blockIdx.x >> 5;           // 4 batches * 32 d-tiles
    const int d0 = (blockIdx.x & 31) * 32;
    const int tid  = threadIdx.x;
    const int lane = tid & 31;
    const int n    = tid & 15;
    const int g    = tid >> 4;                // 0..31 : d-group within block
    const int d    = d0 + g;

    const float a   = -__expf(A_log[(size_t)d * DSTATE + n]);
    const float wdt = W_dt[d];
    const float bdt = b_dt[d];
    float state = 0.f;

    const float* xrow  = xT + ((size_t)b * DMODEL + d) * LSEQ;
    const float* x0row = xT + ((size_t)b * DMODEL + 0) * LSEQ;
    const float* bcb   = BC + (size_t)b * LSEQ * 32;

    for (int lc = 0; lc < LSEQ / 16; ++lc) {
        // lane j preloads step lc*16+j of its d and computes dt, u = dt*x once
        const float xv  = xrow[lc * 16 + n];
        const float x0v = x0row[lc * 16 + n];
        const float z   = fmaf(x0v, wdt, bdt);
        const float dtj = (z > 20.f) ? z : __logf(1.f + __expf(z));
        const float uj  = dtj * xv;
#pragma unroll
        for (int j = 0; j < 16; ++j) {
            const int l = lc * 16 + j;
            const float dtb = __shfl(dtj, (lane & 16) | j, 32);
            const float ub  = __shfl(uj,  (lane & 16) | j, 32);
            const float raw = bcb[(size_t)l * 32 + lane];      // e = lane
            const float Bv  = __shfl(raw, n, 32);
            const float Cv  = __shfl(raw, n + 16, 32);
            const float dA  = __expf(a * dtb);
            state = fmaf(dA, state, ub * Bv);
            float p = state * Cv;
            p += __shfl_xor(p, 8); p += __shfl_xor(p, 4);
            p += __shfl_xor(p, 2); p += __shfl_xor(p, 1);
            if (n == 0) ytile[j][g] = p;
        }
        __syncthreads();
        {   // coalesced bf16 flush: 16 rows x 32 d per chunk
            const int j2 = tid >> 5, dl = tid & 31;
            const int l  = lc * 16 + j2;
            y_bf[((size_t)b * LSEQ + l) * DMODEL + d0 + dl] = f2bf(ytile[j2][dl]);
        }
        __syncthreads();
    }
}

// ---------------------------------------------------------------------------
// Kernel 5: out = y @ W_out^T + b_out via V_WMMA_F32_16X16X32_BF16.
// Block = 8 waves; tile 256(M) x 64(N); each wave: 32x64 via 2 A-fragments x
// 4 B-fragments = 8 accumulators (A reused 4x, B reused 2x -> 1.5 b128 loads
// per WMMA from L2-resident operands).
//   A: lane row = M, halves take 16 contiguous K   (matches B)
//   B: lane col = N, halves take 16 contiguous K   (same k order as A)
// ---------------------------------------------------------------------------
__global__ __launch_bounds__(256)
void k_outgemm(const unsigned short* __restrict__ y_bf,
               const unsigned short* __restrict__ w_bf,
               const float* __restrict__ b_out, float* __restrict__ out) {
    const int tid  = threadIdx.x;
    const int wave = tid >> 5;
    const int lane = tid & 31;
    const int lm   = lane & 15;
    const int half = lane >> 4;

    const int m_base = blockIdx.x * 256 + wave * 32;
    const int o_base = blockIdx.y * 64;

    v8f acc0[4] = {};
    v8f acc1[4] = {};

    const unsigned short* arow0 = y_bf + (size_t)(m_base + lm) * DMODEL + half * 16;
    const unsigned short* arow1 = arow0 + (size_t)16 * DMODEL;
#pragma unroll 2
    for (int k0 = 0; k0 < DMODEL; k0 += 32) {
        v8u a0lo = *(const v8u*)(arow0 + k0);
        v8u a0hi = *(const v8u*)(arow0 + k0 + 8);
        v8u a1lo = *(const v8u*)(arow1 + k0);
        v8u a1hi = *(const v8u*)(arow1 + k0 + 8);
        v16u ar0, ar1;
#pragma unroll
        for (int i = 0; i < 8; ++i) {
            ar0[i] = a0lo[i]; ar0[i + 8] = a0hi[i];
            ar1[i] = a1lo[i]; ar1[i + 8] = a1hi[i];
        }
        v16bf af0 = __builtin_bit_cast(v16bf, ar0);
        v16bf af1 = __builtin_bit_cast(v16bf, ar1);
#pragma unroll
        for (int f = 0; f < 4; ++f) {
            const unsigned short* brow =
                w_bf + (size_t)(o_base + f * 16 + lm) * DMODEL + k0 + half * 16;
            v16bf bfrag = __builtin_bit_cast(v16bf, *(const v16u*)brow);
            acc0[f] = __builtin_amdgcn_wmma_f32_16x16x32_bf16(
                false, af0, false, bfrag, (short)0, acc0[f], false, false);
            acc1[f] = __builtin_amdgcn_wmma_f32_16x16x32_bf16(
                false, af1, false, bfrag, (short)0, acc1[f], false, false);
        }
    }

#pragma unroll
    for (int f = 0; f < 4; ++f) {
        const int col = o_base + f * 16 + lm;
        const float bo = b_out[col];
#pragma unroll
        for (int r = 0; r < 8; ++r) {
            const int row0 = m_base + r + half * 8;
            out[(size_t)row0 * DOUT + col]        = acc0[f][r] + bo;
            out[(size_t)(row0 + 16) * DOUT + col] = acc1[f][r] + bo;
        }
    }
}

// ---------------------------------------------------------------------------
extern "C" void kernel_launch(void* const* d_in, const int* in_sizes, int n_in,
                              void* d_out, int out_size, void* d_ws, size_t ws_size,
                              hipStream_t stream) {
    const float* x     = (const float*)d_in[0];
    const float* A_log = (const float*)d_in[1];
    const float* W_dt  = (const float*)d_in[2];
    const float* b_dt  = (const float*)d_in[3];
    const float* W_xp  = (const float*)d_in[4];
    const float* b_xp  = (const float*)d_in[5];
    const float* W_out = (const float*)d_in[6];
    const float* b_out = (const float*)d_in[7];
    float* out = (float*)d_out;

    char* ws = (char*)d_ws;
    float* xT = (float*)ws;                                  ws += (size_t)B_SZ * DMODEL * LSEQ * 4;
    float* BC = (float*)ws;                                  ws += (size_t)B_SZ * LSEQ * 32 * 4;
    unsigned short* y_bf = (unsigned short*)ws;              ws += (size_t)B_SZ * LSEQ * DMODEL * 2;
    unsigned short* w_bf = (unsigned short*)ws;

    k_transpose<<<dim3(DMODEL / 32, LSEQ / 32, B_SZ), dim3(32, 8), 0, stream>>>(x, xT);
    k_bcproj<<<B_SZ * LSEQ, 256, 0, stream>>>(x, W_xp, b_xp, BC);
    k_wconv<<<(DOUT * DMODEL + 255) / 256, 256, 0, stream>>>(W_out, w_bf, DOUT * DMODEL);
    k_scan<<<B_SZ * (DMODEL / 32), 512, 0, stream>>>(xT, BC, A_log, W_dt, b_dt, y_bf);
    k_outgemm<<<dim3(B_SZ * LSEQ / 256, DOUT / 64), 256, 0, stream>>>(y_bf, w_bf, b_out, out);
}
